// LieMoE_54503134986832
// MI455X (gfx1250) — matrix operand order repaired and checked by reading, compiled, not run on hardware
//
#include <hip/hip_runtime.h>
#include <hip/hip_bf16.h>

// MoE top-4-of-12, sparse (token-gather per expert), f16 WMMA matrix cores.
// D=1024 H=4096 O=1024 E=12 K=4 N=8192.

typedef _Float16 f16;
typedef __attribute__((ext_vector_type(4)))  _Float16 v4h;
typedef __attribute__((ext_vector_type(8)))  _Float16 v8h;
typedef __attribute__((ext_vector_type(16))) _Float16 v16h;
typedef __attribute__((ext_vector_type(8)))  float    v8f;

#define D_DIM 1024
#define H_DIM 4096
#define O_DIM 1024
#define E_NUM 12
#define K_TOP 4
#define N_TOK 8192

#define BM 128
#define BN 128
#define BK 32
#define LDA 40   // halves: 32 + 8 pad -> 80B row stride, 16B aligned, conflict-free
#define LDB 40

#define NK (N_TOK * K_TOP)               // 32768 token-expert pairs
#define ROWS_MAX (NK + E_NUM * BM)       // 34304 (worst-case 128-alignment pad)

// ---- meta layout (int indices into ws base): [0..11]=counts, [16..28]=offsets,
// ---- [32..43]=cursor, [48]=total padded rows.

__global__ __launch_bounds__(256) void init_kernel(int* __restrict__ meta,
                                                   int* __restrict__ token_list) {
  const int g = blockIdx.x * 256 + threadIdx.x;   // grid covers exactly ROWS_MAX
  token_list[g] = 0;
  if (g < E_NUM) meta[g] = 0;
}

__global__ __launch_bounds__(256) void gate_kernel(
    const float* __restrict__ x, const float* __restrict__ gate_w,
    const float* __restrict__ gate_b, float* __restrict__ wts,
    int* __restrict__ idxb, int* __restrict__ meta) {
  const int t = threadIdx.x;
  const int lane = t & 31;
  const int n = blockIdx.x * 8 + (t >> 5);        // one wave32 per token
  float s[E_NUM];
#pragma unroll
  for (int j = 0; j < E_NUM; ++j) s[j] = 0.0f;
  const float* xr = x + (size_t)n * D_DIM;
  for (int d = lane; d < D_DIM; d += 32) {
    const float xv = xr[d];
    const float* g = gate_w + (size_t)d * E_NUM;
#pragma unroll
    for (int j = 0; j < E_NUM; ++j) s[j] = fmaf(xv, g[j], s[j]);
  }
#pragma unroll
  for (int j = 0; j < E_NUM; ++j) {
    s[j] += __shfl_xor(s[j], 16, 32);
    s[j] += __shfl_xor(s[j], 8, 32);
    s[j] += __shfl_xor(s[j], 4, 32);
    s[j] += __shfl_xor(s[j], 2, 32);
    s[j] += __shfl_xor(s[j], 1, 32);
    s[j] += gate_b[j];
  }
  if (lane == 0) {
    float mx = s[0];
#pragma unroll
    for (int j = 1; j < E_NUM; ++j) mx = fmaxf(mx, s[j]);
    float p[E_NUM], Z = 0.0f;
#pragma unroll
    for (int j = 0; j < E_NUM; ++j) { p[j] = __expf(s[j] - mx); Z += p[j]; }
    bool sel[E_NUM];
#pragma unroll
    for (int j = 0; j < E_NUM; ++j) sel[j] = false;
    float kept = 0.0f;
    int ids[K_TOP];
#pragma unroll
    for (int k = 0; k < K_TOP; ++k) {
      int bi = 0; float bv = -3.4e38f;
#pragma unroll
      for (int j = 0; j < E_NUM; ++j)
        if (!sel[j] && s[j] > bv) { bv = s[j]; bi = j; }   // ties: lowest index, like top_k
      sel[bi] = true; ids[k] = bi; kept += p[bi];
    }
    const float denom = kept / Z + 1e-8f;  // sum(masked softmax) + 1e-8
#pragma unroll
    for (int k = 0; k < K_TOP; ++k) {
      wts[n * K_TOP + k] = (p[ids[k]] / Z) / denom;
      idxb[n * K_TOP + k] = ids[k];
      atomicAdd(&meta[ids[k]], 1);
    }
  }
}

__global__ void scan_kernel(int* __restrict__ meta) {
  if (threadIdx.x == 0) {
    int off = 0;
    meta[16] = 0;
    for (int e = 0; e < E_NUM; ++e) {
      const int c = meta[e];
      off += ((c + BM - 1) / BM) * BM;   // 128-align segments -> single-expert tiles
      meta[16 + e + 1] = off;
      meta[32 + e] = 0;                  // cursor
    }
    meta[48] = off;                      // total padded rows
  }
}

__global__ __launch_bounds__(256) void scatter_kernel(
    const int* __restrict__ idxb, int* __restrict__ meta,
    int* __restrict__ token_list, int* __restrict__ pos_of) {
  const int g = blockIdx.x * 256 + threadIdx.x;   // grid covers exactly NK
  const int e = idxb[g];
  const int p = atomicAdd(&meta[32 + e], 1);
  const int row = meta[16 + e] + p;
  token_list[row] = g >> 2;
  pos_of[g] = row;
}

// ---------------- GEMM1: h = relu(gather(x) @ w1[e] + b1[e]) -> f16 ----------------
__global__ __launch_bounds__(256) void gemm1_kernel(
    const float* __restrict__ x, const float* __restrict__ w1,
    const float* __restrict__ b1, const int* __restrict__ meta,
    const int* __restrict__ token_list, f16* __restrict__ h_ws) {
  __shared__ f16 Al[BM * LDA];
  __shared__ f16 Bl[BN * LDB];
  const int total = meta[48];
  const int r0 = blockIdx.y * BM;
  if (r0 >= total) return;                         // uniform exit (EXEC all-1 for WMMA)
  const int* offsets = meta + 16;
  int e = 0;
  while (e < E_NUM - 1 && r0 >= offsets[e + 1]) ++e;
  const int h0 = blockIdx.x * BN;
  const float* __restrict__ w1e = w1 + (size_t)e * D_DIM * H_DIM;
  const float* __restrict__ b1e = b1 + (size_t)e * H_DIM;

  const int t = threadIdx.x;
  const int lane = t & 31;
  const int wm = (t >> 5) >> 2;                    // 0..1
  const int wn = (t >> 5) & 3;                     // 0..3
  const int l15 = lane & 15;
  const int kbA = (lane < 16) ? 0 : 8;             // ISA 16-bit A layout
  const int kbB = (lane < 16) ? 0 : 8;             // ISA 16-bit B layout

  int tokRow[4];
#pragma unroll
  for (int p = 0; p < 4; ++p) tokRow[p] = token_list[r0 + p * 32 + (t >> 3)];

  v8f acc[4][2];
#pragma unroll
  for (int mi = 0; mi < 4; ++mi)
#pragma unroll
    for (int ni = 0; ni < 2; ++ni) acc[mi][ni] = (v8f){0,0,0,0,0,0,0,0};

  for (int kt = 0; kt < D_DIM / BK; ++kt) {
    const int kd0 = kt * BK;
    float4 av[4];
#pragma unroll
    for (int p = 0; p < 4; ++p)
      av[p] = *(const float4*)(x + (size_t)tokRow[p] * D_DIM + kd0 + (t & 7) * 4);
    float4 bv[4];
#pragma unroll
    for (int p = 0; p < 4; ++p) {
      const int kd = p * 8 + (t >> 5);
      bv[p] = *(const float4*)(w1e + (size_t)(kd0 + kd) * H_DIM + h0 + (t & 31) * 4);
    }
    __syncthreads();
#pragma unroll
    for (int p = 0; p < 4; ++p) {
      const int row = p * 32 + (t >> 3);
      v4h q = {(f16)av[p].x, (f16)av[p].y, (f16)av[p].z, (f16)av[p].w};
      *(v4h*)(Al + row * LDA + (t & 7) * 4) = q;
    }
#pragma unroll
    for (int p = 0; p < 4; ++p) {                  // transpose B into [n][k]
      const int kd = p * 8 + (t >> 5);
      const int hb = (t & 31) * 4;
      Bl[(hb + 0) * LDB + kd] = (f16)bv[p].x;
      Bl[(hb + 1) * LDB + kd] = (f16)bv[p].y;
      Bl[(hb + 2) * LDB + kd] = (f16)bv[p].z;
      Bl[(hb + 3) * LDB + kd] = (f16)bv[p].w;
    }
    __syncthreads();

    v16h afr[4];
#pragma unroll
    for (int mi = 0; mi < 4; ++mi) {
      const int m = wm * 64 + mi * 16 + l15;
      v8h lo = *(const v8h*)(Al + m * LDA + kbA);        // K = kb..kb+7
      v8h hi = *(const v8h*)(Al + m * LDA + kbA + 16);   // K = kb+16..kb+23
      afr[mi] = __builtin_shufflevector(lo, hi, 0,1,2,3,4,5,6,7,8,9,10,11,12,13,14,15);
    }
    v16h bfr[2];
#pragma unroll
    for (int ni = 0; ni < 2; ++ni) {
      const int c = wn * 32 + ni * 16 + l15;
      v8h lo = *(const v8h*)(Bl + c * LDB + kbB);        // K = kb..kb+7
      v8h hi = *(const v8h*)(Bl + c * LDB + kbB + 16);   // K = kb+16..kb+23
      bfr[ni] = __builtin_shufflevector(lo, hi, 0,1,2,3,4,5,6,7,8,9,10,11,12,13,14,15);
    }
#pragma unroll
    for (int mi = 0; mi < 4; ++mi)
#pragma unroll
      for (int ni = 0; ni < 2; ++ni)
        acc[mi][ni] = __builtin_amdgcn_wmma_f32_16x16x32_f16(
            false, afr[mi], false, bfr[ni], (short)0, acc[mi][ni], false, false);
  }

#pragma unroll
  for (int mi = 0; mi < 4; ++mi)
#pragma unroll
    for (int ni = 0; ni < 2; ++ni) {
      const int col = h0 + wn * 32 + ni * 16 + l15;
      const float bb = b1e[col];
      const int mb = r0 + wm * 64 + mi * 16 + ((lane >> 4) << 3);
#pragma unroll
      for (int r = 0; r < 8; ++r) {
        float v = acc[mi][ni][r] + bb;
        v = v > 0.0f ? v : 0.0f;
        h_ws[(size_t)(mb + r) * H_DIM + col] = (f16)v;
      }
    }
}

// ---------------- GEMM2: out_rows = h @ w2[e] + b2[e] (f32) ----------------
__global__ __launch_bounds__(256) void gemm2_kernel(
    const f16* __restrict__ h_ws, const float* __restrict__ w2,
    const float* __restrict__ b2, const int* __restrict__ meta,
    float* __restrict__ out_rows) {
  __shared__ f16 Al[BM * LDA];
  __shared__ f16 Bl[BN * LDB];
  const int total = meta[48];
  const int r0 = blockIdx.y * BM;
  if (r0 >= total) return;
  const int* offsets = meta + 16;
  int e = 0;
  while (e < E_NUM - 1 && r0 >= offsets[e + 1]) ++e;
  const int o0 = blockIdx.x * BN;
  const float* __restrict__ w2e = w2 + (size_t)e * H_DIM * O_DIM;
  const float* __restrict__ b2e = b2 + (size_t)e * O_DIM;

  const int t = threadIdx.x;
  const int lane = t & 31;
  const int wm = (t >> 5) >> 2;
  const int wn = (t >> 5) & 3;
  const int l15 = lane & 15;
  const int kbA = (lane < 16) ? 0 : 8;
  const int kbB = (lane < 16) ? 0 : 8;

  v8f acc[4][2];
#pragma unroll
  for (int mi = 0; mi < 4; ++mi)
#pragma unroll
    for (int ni = 0; ni < 2; ++ni) acc[mi][ni] = (v8f){0,0,0,0,0,0,0,0};

  for (int kt = 0; kt < H_DIM / BK; ++kt) {
    const int kd0 = kt * BK;
    v8h av[2];
#pragma unroll
    for (int p = 0; p < 2; ++p) {
      const int row = p * 64 + (t >> 2);
      av[p] = *(const v8h*)(h_ws + (size_t)(r0 + row) * H_DIM + kd0 + (t & 3) * 8);
    }
    float4 bv[4];
#pragma unroll
    for (int p = 0; p < 4; ++p) {
      const int kd = p * 8 + (t >> 5);
      bv[p] = *(const float4*)(w2e + (size_t)(kd0 + kd) * O_DIM + o0 + (t & 31) * 4);
    }
    __syncthreads();
#pragma unroll
    for (int p = 0; p < 2; ++p) {
      const int row = p * 64 + (t >> 2);
      *(v8h*)(Al + row * LDA + (t & 3) * 8) = av[p];
    }
#pragma unroll
    for (int p = 0; p < 4; ++p) {
      const int kd = p * 8 + (t >> 5);
      const int ob = (t & 31) * 4;
      Bl[(ob + 0) * LDB + kd] = (f16)bv[p].x;
      Bl[(ob + 1) * LDB + kd] = (f16)bv[p].y;
      Bl[(ob + 2) * LDB + kd] = (f16)bv[p].z;
      Bl[(ob + 3) * LDB + kd] = (f16)bv[p].w;
    }
    __syncthreads();

    v16h afr[4];
#pragma unroll
    for (int mi = 0; mi < 4; ++mi) {
      const int m = wm * 64 + mi * 16 + l15;
      v8h lo = *(const v8h*)(Al + m * LDA + kbA);
      v8h hi = *(const v8h*)(Al + m * LDA + kbA + 16);
      afr[mi] = __builtin_shufflevector(lo, hi, 0,1,2,3,4,5,6,7,8,9,10,11,12,13,14,15);
    }
    v16h bfr[2];
#pragma unroll
    for (int ni = 0; ni < 2; ++ni) {
      const int c = wn * 32 + ni * 16 + l15;
      v8h lo = *(const v8h*)(Bl + c * LDB + kbB);
      v8h hi = *(const v8h*)(Bl + c * LDB + kbB + 16);
      bfr[ni] = __builtin_shufflevector(lo, hi, 0,1,2,3,4,5,6,7,8,9,10,11,12,13,14,15);
    }
#pragma unroll
    for (int mi = 0; mi < 4; ++mi)
#pragma unroll
      for (int ni = 0; ni < 2; ++ni)
        acc[mi][ni] = __builtin_amdgcn_wmma_f32_16x16x32_f16(
            false, afr[mi], false, bfr[ni], (short)0, acc[mi][ni], false, false);
  }

#pragma unroll
  for (int mi = 0; mi < 4; ++mi)
#pragma unroll
    for (int ni = 0; ni < 2; ++ni) {
      const int col = o0 + wn * 32 + ni * 16 + l15;
      const float bb = b2e[col];
      const int mb = r0 + wm * 64 + mi * 16 + ((lane >> 4) << 3);
#pragma unroll
      for (int r = 0; r < 8; ++r)
        out_rows[(size_t)(mb + r) * O_DIM + col] = acc[mi][ni][r] + bb;
    }
}

__global__ __launch_bounds__(256) void reduce_kernel(
    const float* __restrict__ out_rows, const float* __restrict__ wts,
    const int* __restrict__ pos_of, float* __restrict__ out) {
  const int g = blockIdx.x * 256 + threadIdx.x;
  const int n = g >> 8;
  const int o4 = (g & 255) * 4;
  float ax = 0, ay = 0, az = 0, aw = 0;
#pragma unroll
  for (int k = 0; k < K_TOP; ++k) {
    const float w = wts[n * K_TOP + k];
    const int pos = pos_of[n * K_TOP + k];
    const float4 v = *(const float4*)(out_rows + (size_t)pos * O_DIM + o4);
    ax = fmaf(w, v.x, ax); ay = fmaf(w, v.y, ay);
    az = fmaf(w, v.z, az); aw = fmaf(w, v.w, aw);
  }
  float4 r; r.x = ax; r.y = ay; r.z = az; r.w = aw;
  *(float4*)(out + (size_t)n * O_DIM + o4) = r;
}

extern "C" void kernel_launch(void* const* d_in, const int* in_sizes, int n_in,
                              void* d_out, int out_size, void* d_ws, size_t ws_size,
                              hipStream_t stream) {
  const float* x      = (const float*)d_in[0];
  const float* gate_w = (const float*)d_in[1];
  const float* gate_b = (const float*)d_in[2];
  const float* w1     = (const float*)d_in[3];
  const float* b1     = (const float*)d_in[4];
  const float* w2     = (const float*)d_in[5];
  const float* b2     = (const float*)d_in[6];
  float* out = (float*)d_out;

  // Workspace carve-up (~403 MiB total): meta | weights | idx | pos_of | token_list | h(f16) | out_rows(f32)
  char* ws = (char*)d_ws;
  int*   meta       = (int*)ws;
  float* wts        = (float*)(ws + 256);
  int*   idxb       = (int*)(ws + 256 + (size_t)NK * 4);
  int*   pos_of     = (int*)(ws + 256 + (size_t)NK * 8);
  int*   token_list = (int*)(ws + 256 + (size_t)NK * 12);
  size_t o5 = 256 + (size_t)NK * 12 + (size_t)ROWS_MAX * 4;
  o5 = (o5 + 255) & ~(size_t)255;
  f16* h_ws = (f16*)(ws + o5);
  size_t o6 = o5 + (size_t)ROWS_MAX * H_DIM * 2;
  o6 = (o6 + 255) & ~(size_t)255;
  float* out_rows = (float*)(ws + o6);

  init_kernel<<<ROWS_MAX / 256, 256, 0, stream>>>(meta, token_list);
  gate_kernel<<<N_TOK / 8, 256, 0, stream>>>(x, gate_w, gate_b, wts, idxb, meta);
  scan_kernel<<<1, 32, 0, stream>>>(meta);
  scatter_kernel<<<NK / 256, 256, 0, stream>>>(idxb, meta, token_list, pos_of);
  dim3 g1(H_DIM / BN, ROWS_MAX / BM);
  gemm1_kernel<<<g1, 256, 0, stream>>>(x, w1, b1, meta, token_list, h_ws);
  dim3 g2(O_DIM / BN, ROWS_MAX / BM);
  gemm2_kernel<<<g2, 256, 0, stream>>>(h_ws, w2, b2, meta, out_rows);
  reduce_kernel<<<(N_TOK * O_DIM / 4) / 256, 256, 0, stream>>>(out_rows, wts, pos_of, out);
}
